// REGCN_13314398617723
// MI455X (gfx1250) — matrix operand order repaired and checked by reading, not compiled, not run on hardware
//
#include <hip/hip_runtime.h>
#include <cstdint>

#define HIDN   128
#define OUTN   349
#define OUTP   352           // padded to 22 * 16
#define NNODE  100000
#define NEDGE  500000
#define SCL    100.0f
#define BNEPS  1e-5f
#define NEGHUGE (-3.402823466e38f)

typedef __attribute__((ext_vector_type(16))) __bf16 v16bf;
typedef __attribute__((ext_vector_type(8)))  __bf16 v8bf;
typedef __attribute__((ext_vector_type(8)))  float  v8f;
typedef int v4i __attribute__((vector_size(16)));   // matches async-LDS builtin pointee

__device__ __forceinline__ float lrelu01(float v) { return v > 0.0f ? v : 0.01f * v; }

__device__ __forceinline__ v8f wmma_bf16(v16bf a, v16bf b, v8f c) {
    return __builtin_amdgcn_wmma_f32_16x16x32_bf16(false, a, false, b, (short)0, c, false, false);
}

// ---- A fragment from a 16x128 bf16 LDS tile (ISA 7.12.2 16-bit A 16x32) ----
// lane L<16: row L, K = kk+{0..15}; lane L>=16: row L-16, K = kk+{16..31} (contiguous, matches B)
__device__ __forceinline__ v16bf afrag_lds(const __bf16* tile, int kk, int lane) {
    const int half = lane >> 4;
    const __bf16* p = tile + (lane & 15) * HIDN + kk + half * 16;
    v8bf lo = *(const v8bf*)p;          // 16B -> ds_load_b128
    v8bf hi = *(const v8bf*)(p + 8);    // 16B -> ds_load_b128
    return __builtin_shufflevector(lo, hi, 0, 1, 2, 3, 4, 5, 6, 7,
                                           8, 9, 10, 11, 12, 13, 14, 15);
}

// ---- B fragment from pre-packed WT[N][128] bf16 (transposed weights) ----
// lane holds column n0+(lane&15); elements e=0..15 <-> K = kk + half*16 + e (contiguous!)
__device__ __forceinline__ v16bf bfrag_packed(const __bf16* __restrict__ WT,
                                              int kk, int n0, int lane) {
    const int half = lane >> 4;
    const __bf16* p = WT + (size_t)(n0 + (lane & 15)) * HIDN + kk + half * 16;
    return *(const v16bf*)p;            // 32B contiguous -> 2x global_load_b128
}

// ---- pack row-major f32 W[K=128 x Nvalid] into bf16 WT[Npad][128] (transpose) ----
__global__ void k_pack_wt(const float* __restrict__ W, __bf16* __restrict__ WT,
                          int Nvalid, int Npad) {
    const int i = blockIdx.x * blockDim.x + threadIdx.x;
    if (i >= Npad * HIDN) return;
    const int n = i / HIDN, k = i - n * HIDN;
    WT[i] = (n < Nvalid) ? (__bf16)W[(size_t)k * Nvalid + n] : (__bf16)0.0f;
}

// ---- Stage A: gathered per-type input linears; both type GEMMs + per-row select ----
__global__ void k_group_input(const float* __restrict__ x0, const float* __restrict__ x1,
                              const int* __restrict__ node_type, const int* __restrict__ lidx,
                              const __bf16* __restrict__ WT0, const float* __restrict__ b0,
                              const __bf16* __restrict__ WT1, const float* __restrict__ b1,
                              float* __restrict__ H, __bf16* __restrict__ Hbf) {
    __shared__ __bf16 tile[16 * HIDN];
    const int lane = threadIdx.x & 31;
    const int wv   = threadIdx.x >> 5;
    const int m0   = blockIdx.x * 16;
    {   // gather + convert the 16x128 A tile into LDS (each thread: 8 elements)
        const int r = threadIdx.x >> 4;
        const int s = (threadIdx.x & 15) * 8;
        const int rowi = m0 + r;
        const int t = node_type[rowi];
        const float* src = (t ? x1 : x0) + (size_t)lidx[rowi] * HIDN + s;
        v8bf o;
#pragma unroll
        for (int j = 0; j < 8; ++j) o[j] = (__bf16)src[j];
        *(v8bf*)(tile + r * HIDN + s) = o;
    }
    __syncthreads();
    const int n0   = wv * 16;           // 8 waves cover 128 output cols
    const int half = lane >> 4;
    v8f c0 = {}; v8f c1 = {};
#pragma unroll
    for (int kk = 0; kk < HIDN; kk += 32) {
        v16bf a  = afrag_lds(tile, kk, lane);
        c0 = wmma_bf16(a, bfrag_packed(WT0, kk, n0, lane), c0);
        c1 = wmma_bf16(a, bfrag_packed(WT1, kk, n0, lane), c1);
    }
    const int colc = n0 + (lane & 15);
#pragma unroll
    for (int r = 0; r < 8; ++r) {
        const int rowi = m0 + r + half * 8;
        const int tr = node_type[rowi];
        const float v = tr ? (c1[r] + b1[colc]) : (c0[r] + b0[colc]);
        H[(size_t)rowi * HIDN + colc]   = v;
        Hbf[(size_t)rowi * HIDN + colc] = (__bf16)v;
    }
}

// ---- WMMA GEMM: C[M x N] = Abf[M x 128] @ WT^T, A staged to LDS (async if available) ----
__global__ void k_gemm(const __bf16* __restrict__ Abf, const __bf16* __restrict__ WT,
                       float* __restrict__ C, int ldc, int ntiles) {
    __shared__ __bf16 tile[16 * HIDN];
    const int lane = threadIdx.x & 31;
    const int wv   = threadIdx.x >> 5;
    const int m0   = blockIdx.x * 16;
    {   // stage 16x128 bf16 A tile into LDS; each thread moves one 16B chunk
        const int r = threadIdx.x >> 4;
        const int s = (threadIdx.x & 15) * 8;
        const __bf16* gsrc = Abf + (size_t)(m0 + r) * HIDN + s;
        __bf16* ldst = tile + r * HIDN + s;
#if __has_builtin(__builtin_amdgcn_global_load_async_to_lds_b128)
        // CDNA5 async DMA path: memory -> LDS without VGPR round-trip (ASYNCcnt)
        __builtin_amdgcn_global_load_async_to_lds_b128(
            (__attribute__((address_space(1))) v4i*)(uintptr_t)gsrc,
            (__attribute__((address_space(3))) v4i*)(unsigned)(uintptr_t)ldst,
            0, 0);
#if __has_builtin(__builtin_amdgcn_s_wait_asynccnt)
        __builtin_amdgcn_s_wait_asynccnt(0);
#else
        asm volatile("s_wait_asynccnt 0" ::: "memory");
#endif
#else
        *(v8bf*)ldst = *(const v8bf*)gsrc;
#endif
    }
    __syncthreads();
    const int t = blockIdx.y * 8 + wv;
    if (t >= ntiles) return;            // uniform whole-wave exit, EXEC stays all-ones
    const int n0 = t * 16;
    v8f c = {};
#pragma unroll
    for (int kk = 0; kk < HIDN; kk += 32)
        c = wmma_bf16(afrag_lds(tile, kk, lane), bfrag_packed(WT, kk, n0, lane), c);
    const int half = lane >> 4;
    const int colc = n0 + (lane & 15);
#pragma unroll
    for (int r = 0; r < 8; ++r)
        C[(size_t)(m0 + r + half * 8) * ldc + colc] = c[r];
}

// ---- weighted in-degree at targets, both conv layers in one pass ----
__global__ void k_degree(const int* __restrict__ col, const int* __restrict__ etype,
                         const float* __restrict__ rw0, const float* __restrict__ rw1,
                         float* __restrict__ deg0, float* __restrict__ deg1) {
    const int e = blockIdx.x * blockDim.x + threadIdx.x;
    if (e >= NEDGE) return;
    const int et = etype[e];
    const int c  = col[e];
    atomicAdd(&deg0[c], lrelu01(rw0[et] * SCL));
    atomicAdd(&deg1[c], lrelu01(rw1[et] * SCL));
}

// ---- edge scatter: one wave32 per edge, Out[col] += ew * Xs[row] ----
__global__ void k_scatter(const int* __restrict__ row, const int* __restrict__ col,
                          const int* __restrict__ etype, const float* __restrict__ rw,
                          const float* __restrict__ deg,
                          const float* __restrict__ Xs, int ldx,
                          float* __restrict__ Out, int ldo, int F) {
    const int gw   = (int)((blockIdx.x * (size_t)blockDim.x + threadIdx.x) >> 5);
    const int lane = threadIdx.x & 31;
    if (gw >= NEDGE) return;
    const int r  = row[gw];
    const float* src = Xs + (size_t)r * ldx;
    __builtin_prefetch(src + lane * ((F + 31) >> 5), 0, 1);  // pull row while we divide
    const int c  = col[gw];
    const int et = etype[gw];
    const float w  = lrelu01(rw[et] * SCL);
    const float ew = w / fabsf(deg[c]);
    float* dst = Out + (size_t)c * ldo;
    for (int f = lane; f < F; f += 32)
        atomicAdd(&dst[f], ew * src[f]);
}

// ---- residual + bias, accumulate BN partial sums (32 rows per 128-thread block) ----
__global__ void k_resid_bnstat(const float* __restrict__ H0, const float* __restrict__ b,
                               float* __restrict__ X,
                               float* __restrict__ bnsum, float* __restrict__ bnsq) {
    const int c  = threadIdx.x;          // 0..127
    const int r0 = blockIdx.x * 32;
    const float bias = b[c];
    float s = 0.0f, q = 0.0f;
    for (int r = 0; r < 32; ++r) {
        const size_t i = (size_t)(r0 + r) * HIDN + c;
        const float v = X[i] + bias + H0[i];
        X[i] = v;
        s += v; q += v * v;
    }
    atomicAdd(&bnsum[c], s);
    atomicAdd(&bnsq[c], q);
}

__global__ void k_bn_finalize(const float* __restrict__ bnsum, const float* __restrict__ bnsq,
                              const float* __restrict__ gamma, const float* __restrict__ beta,
                              float* __restrict__ scale, float* __restrict__ shift) {
    const int c = threadIdx.x;
    const float invN = 1.0f / (float)NNODE;
    const float mean = bnsum[c] * invN;
    const float var  = bnsq[c] * invN - mean * mean;
    const float sc   = rsqrtf(var + BNEPS) * gamma[c];
    scale[c] = sc;
    shift[c] = beta[c] - mean * sc;
}

// ---- BN apply + PReLU; also writes the bf16 shadow used as GEMM-1 A operand ----
__global__ void k_bn_prelu(float* __restrict__ X, __bf16* __restrict__ Xbf,
                           const float* __restrict__ scale, const float* __restrict__ shift,
                           const float* __restrict__ pa) {
    const size_t i2 = blockIdx.x * (size_t)blockDim.x + threadIdx.x;
    if (i2 >= (size_t)NNODE * HIDN / 2) return;
    const size_t i = i2 * 2;
    const int c = (int)(i & (HIDN - 1));
    const float a = pa[0];
    float v0 = X[i] * scale[c] + shift[c];
    float v1 = X[i + 1] * scale[c + 1] + shift[c + 1];
    v0 = v0 >= 0.0f ? v0 : a * v0;
    v1 = v1 >= 0.0f ? v1 : a * v1;
    X[i] = v0; X[i + 1] = v1;
    Xbf[i] = (__bf16)v0; Xbf[i + 1] = (__bf16)v1;
}

// ---- fold conv1 bias, then in-place row log-softmax (one 128-thread block per row) ----
__global__ void k_logsoftmax_bias(float* __restrict__ Out, const float* __restrict__ b) {
    __shared__ float red[128];
    float* rowp = Out + (size_t)blockIdx.x * OUTN;
    const int t = threadIdx.x;
    float vals[3];
    float mx = NEGHUGE;
#pragma unroll
    for (int j = 0; j < 3; ++j) {
        const int c = t + j * 128;
        const float v = (c < OUTN) ? rowp[c] + b[c] : NEGHUGE;
        vals[j] = v;
        mx = fmaxf(mx, v);
    }
    red[t] = mx; __syncthreads();
    for (int s = 64; s > 0; s >>= 1) { if (t < s) red[t] = fmaxf(red[t], red[t + s]); __syncthreads(); }
    mx = red[0]; __syncthreads();
    float se = 0.0f;
#pragma unroll
    for (int j = 0; j < 3; ++j) {
        const int c = t + j * 128;
        if (c < OUTN) se += expf(vals[j] - mx);
    }
    red[t] = se; __syncthreads();
    for (int s = 64; s > 0; s >>= 1) { if (t < s) red[t] += red[t + s]; __syncthreads(); }
    const float lse = logf(red[0]) + mx;
#pragma unroll
    for (int j = 0; j < 3; ++j) {
        const int c = t + j * 128;
        if (c < OUTN) rowp[c] = vals[j] - lse;
    }
}

extern "C" void kernel_launch(void* const* d_in, const int* in_sizes, int n_in,
                              void* d_out, int out_size, void* d_ws, size_t ws_size,
                              hipStream_t stream) {
    (void)in_sizes; (void)n_in; (void)out_size; (void)ws_size;

    const float* x0       = (const float*)d_in[0];
    const float* x1       = (const float*)d_in[1];
    const int*   ei       = (const int*)d_in[2];
    const int*   row      = ei;
    const int*   col      = ei + NEDGE;
    const int*   etype    = (const int*)d_in[3];
    const int*   ntype    = (const int*)d_in[4];
    const int*   lidx     = (const int*)d_in[5];
    const float* lin0_w   = (const float*)d_in[6];
    const float* lin0_b   = (const float*)d_in[7];
    const float* lin1_w   = (const float*)d_in[8];
    const float* lin1_b   = (const float*)d_in[9];
    const float* conv0_w  = (const float*)d_in[10];
    const float* conv0_b  = (const float*)d_in[11];
    const float* conv0_rw = (const float*)d_in[12];
    const float* conv1_w  = (const float*)d_in[13];
    const float* conv1_b  = (const float*)d_in[14];
    const float* conv1_rw = (const float*)d_in[15];
    const float* gamma    = (const float*)d_in[16];
    const float* beta     = (const float*)d_in[17];
    const float* pa       = (const float*)d_in[18];
    float* out = (float*)d_out;

    // ---- workspace carve-up: f32 regions first, then bf16 regions ----
    float* fws = (float*)d_ws;
    float* H0   = fws;                                // 100000*128
    float* X0s  = H0  + (size_t)NNODE * HIDN;         // 100000*128
    float* ACC  = X0s + (size_t)NNODE * HIDN;         // 100000*128 (conv0 acc -> H1)
    float* X1s  = ACC + (size_t)NNODE * HIDN;         // 100000*352
    float* deg0 = X1s + (size_t)NNODE * OUTP;         // 100000
    float* deg1 = deg0 + NNODE;                       // 100000
    float* bns  = deg1 + NNODE;                       // 128
    float* bnq  = bns + 128;
    float* bsc  = bnq + 128;
    float* bsh  = bsc + 128;
    __bf16* bws  = (__bf16*)(bsh + 128);
    __bf16* H0bf  = bws;                              // 100000*128 bf16
    __bf16* ACCbf = H0bf  + (size_t)NNODE * HIDN;     // 100000*128 bf16
    __bf16* WTl0  = ACCbf + (size_t)NNODE * HIDN;     // 128*128
    __bf16* WTl1  = WTl0 + HIDN * HIDN;               // 128*128
    __bf16* WTc0  = WTl1 + HIDN * HIDN;               // 128*128
    __bf16* WTc1  = WTc0 + HIDN * HIDN;               // 352*128

    // deterministic per-call re-init of all accumulators (graph-capture safe)
    (void)hipMemsetAsync(ACC,  0, (size_t)NNODE * HIDN * sizeof(float), stream);
    (void)hipMemsetAsync(deg0, 0, (size_t)(2 * NNODE + 512) * sizeof(float), stream);
    (void)hipMemsetAsync(out,  0, (size_t)NNODE * OUTN * sizeof(float), stream);

    // pack all weight matrices into bf16 fragment-friendly (transposed) layout
    k_pack_wt<<<(HIDN * HIDN + 255) / 256, 256, 0, stream>>>(lin0_w,  WTl0, HIDN, HIDN);
    k_pack_wt<<<(HIDN * HIDN + 255) / 256, 256, 0, stream>>>(lin1_w,  WTl1, HIDN, HIDN);
    k_pack_wt<<<(HIDN * HIDN + 255) / 256, 256, 0, stream>>>(conv0_w, WTc0, HIDN, HIDN);
    k_pack_wt<<<(OUTP * HIDN + 255) / 256, 256, 0, stream>>>(conv1_w, WTc1, OUTN, OUTP);

    const int rowTiles = NNODE / 16;                  // 6250

    k_group_input<<<rowTiles, 256, 0, stream>>>(x0, x1, ntype, lidx,
                                                WTl0, lin0_b, WTl1, lin1_b, H0, H0bf);
    k_gemm<<<dim3(rowTiles, 1), 256, 0, stream>>>(H0bf, WTc0, X0s, HIDN, 8);
    k_degree<<<(NEDGE + 255) / 256, 256, 0, stream>>>(col, etype, conv0_rw, conv1_rw, deg0, deg1);
    k_scatter<<<NEDGE / 8, 256, 0, stream>>>(row, col, etype, conv0_rw, deg0,
                                             X0s, HIDN, ACC, HIDN, HIDN);
    k_resid_bnstat<<<NNODE / 32, 128, 0, stream>>>(H0, conv0_b, ACC, bns, bnq);
    k_bn_finalize<<<1, 128, 0, stream>>>(bns, bnq, gamma, beta, bsc, bsh);
    k_bn_prelu<<<(NNODE * HIDN / 2 + 255) / 256, 256, 0, stream>>>(ACC, ACCbf, bsc, bsh, pa);
    k_gemm<<<dim3(rowTiles, 3), 256, 0, stream>>>(ACCbf, WTc1, X1s, OUTP, OUTP / 16);
    k_scatter<<<NEDGE / 8, 256, 0, stream>>>(row, col, etype, conv1_rw, deg1,
                                             X1s, OUTP, out, OUTN, OUTN);
    k_logsoftmax_bias<<<NNODE, 128, 0, stream>>>(out, conv1_b);
}
